// EdgeDecoder_35098472742971
// MI455X (gfx1250) — compile-verified, matrix operand order checked by reading, not run on hardware
//
#include <hip/hip_runtime.h>

// ---------------------------------------------------------------------------
// EdgeDecoder for MI455X (gfx1250, wave32, WMMA)
//   x  = z[e0] * z[e1]                  [E,128]  (bf16 after product)
//   h  = mish(x @ W1^T + b1)            [E,256]  (bf16 WMMA, f32 accum)
//   o  = sigmoid(h @ w2 + b2)           [E,1]    (VALU reduction)
// ---------------------------------------------------------------------------

#define N_NODES 100000
#define N_EDGES 500000
#define D_IN    128
#define D_HID   256

typedef __attribute__((ext_vector_type(16))) __bf16 v16bf;
typedef __attribute__((ext_vector_type(8)))  float  v8f;

union BFrag {
    uint4  q[2];   // 2 x 16B = 16 bf16 values
    v16bf  v;
};

// fp32 pair -> packed bf16 (RNE). Plain casts let the backend select the
// native bf16 convert (v_cvt_pk_bf16_f32 class) when available.
__device__ __forceinline__ unsigned int pack_bf16x2(float x, float y) {
    union { __bf16 h[2]; unsigned int u; } p;
    p.h[0] = (__bf16)x;
    p.h[1] = (__bf16)y;
    return p.u;
}

// Branch-free mish: with e = exp(x), tanh(softplus(x)) = e(e+2) / (e(e+2)+2).
// Clamp exp arg at 40 (ratio == 1.0f in fp32 for x >= ~20, so clamp is exact).
__device__ __forceinline__ float mish_f(float x) {
    float e = __expf(fminf(x, 40.0f));
    float n = e * (e + 2.0f);
    return x * __fdividef(n, n + 2.0f);
}

// --- prep: W1 (256x128 f32, row-major) -> bf16 copy in workspace ------------
__global__ void w1_to_bf16_kernel(const float* __restrict__ W1,
                                  unsigned int* __restrict__ w1bf_u32) {
    int i = blockIdx.x * blockDim.x + threadIdx.x;   // pair index
    if (i < (D_HID * D_IN) / 2)
        w1bf_u32[i] = pack_bf16x2(W1[2 * i], W1[2 * i + 1]);
}

// --- main fused kernel: one 16-edge tile per 128-thread block ---------------
__global__ __launch_bounds__(128)
void edge_decoder_kernel(const float* __restrict__ z,
                         const int*   __restrict__ edge,    // [2, E]
                         const unsigned short* __restrict__ w1bf, // [256,128] bf16
                         const float* __restrict__ b1,      // [256]
                         const float* __restrict__ W2,      // [256] (1x256)
                         const float* __restrict__ b2,      // [1]
                         float*       __restrict__ out) {   // [E]
    // A tile: 16 rows x 128 cols bf16, padded row stride 136 (bank spread)
    __shared__ __align__(16) unsigned short aTile[16 * 136];
    __shared__ float partRed[16];

    const int ebase = blockIdx.x * 16;
    const int tid   = threadIdx.x;

    if (tid < 16) partRed[tid] = 0.0f;

    // ---------------- Phase 1: gather + multiply + bf16 convert -------------
    // 16 rows x 32 float4 = 512 tasks over 128 threads (4 each)
    const float4* z4 = (const float4*)z;
    #pragma unroll
    for (int i = 0; i < 4; ++i) {
        int task = tid + i * 128;
        int row  = task >> 5;          // 0..15
        int c4   = task & 31;          // 0..31 (float4 column)
        int e    = ebase + row;
        int n0   = edge[e];
        int n1   = edge[N_EDGES + e];
        float4 a = z4[n0 * 32 + c4];
        float4 b = z4[n1 * 32 + c4];
        unsigned int lo = pack_bf16x2(a.x * b.x, a.y * b.y);
        unsigned int hi = pack_bf16x2(a.z * b.z, a.w * b.w);
        *(uint2*)&aTile[row * 136 + c4 * 4] = make_uint2(lo, hi);
    }
    __syncthreads();

    // ---------------- Phase 2: WMMA GEMM1 + mish + dot(w2) ------------------
    const int wave    = tid >> 5;
    const int lane    = tid & 31;
    const int m16     = lane & 15;
    const int halfSel = lane >> 4;         // 0: lanes 0-15, 1: lanes 16-31
    const int halfOff = halfSel * 8;       // K sub-offset per ISA A/B layout

    // A fragments (4 K-chunks of 32), per 16-bit A 16x32 VGPR layout:
    //   lanes 0-15:  K = c*32 + {0..7, 16..23}
    //   lanes 16-31: K = c*32 + {8..15, 24..31}
    BFrag afrag[4];
    #pragma unroll
    for (int c = 0; c < 4; ++c) {
        int lo = m16 * 136 + 32 * c + halfOff;
        afrag[c].q[0] = *(const uint4*)&aTile[lo];        // K lo .. lo+7
        afrag[c].q[1] = *(const uint4*)&aTile[lo + 16];   // K lo+16 .. lo+23
    }

    float s[8];
    #pragma unroll
    for (int r = 0; r < 8; ++r) s[r] = 0.0f;

    #pragma unroll
    for (int t = 0; t < 4; ++t) {
        const int nbase = wave * 64 + t * 16;
        // B[k][n] = W1[nbase+n][k] -> per-lane contiguous-K reads of W1 row
        const unsigned short* wrow = w1bf + (nbase + m16) * D_IN;
        v8f acc = {};
        #pragma unroll
        for (int c = 0; c < 4; ++c) {
            BFrag bfrag;
            int lo = 32 * c + halfOff;
            bfrag.q[0] = *(const uint4*)(wrow + lo);
            bfrag.q[1] = *(const uint4*)(wrow + lo + 16);
            acc = __builtin_amdgcn_wmma_f32_16x16x32_bf16(
                      /*neg_a=*/false, afrag[c].v,
                      /*neg_b=*/false, bfrag.v,
                      /*c_mod=*/(short)0, acc,
                      /*reuse_a=*/false, /*reuse_b=*/false);
        }
        // C layout: VGPR r -> row (r + 8*halfSel), col N = nbase + m16
        const int   ncol = nbase + m16;
        const float b1v  = b1[ncol];
        const float w2v  = W2[ncol];
        #pragma unroll
        for (int r = 0; r < 8; ++r) {
            float x = acc[r] + b1v;
            s[r] += mish_f(x) * w2v;
        }
    }

    // reduce over the 16 lanes of each half (each half owns rows r / r+8)
    #pragma unroll
    for (int m = 8; m >= 1; m >>= 1) {
        #pragma unroll
        for (int r = 0; r < 8; ++r) s[r] += __shfl_xor(s[r], m, 32);
    }
    if (m16 == 0) {
        #pragma unroll
        for (int r = 0; r < 8; ++r)
            atomicAdd(&partRed[r + halfSel * 8], s[r]);   // ds_add_f32
    }
    __syncthreads();

    // ---------------- Phase 3: sigmoid + store ------------------------------
    if (tid < 16) {
        float v = partRed[tid] + b2[0];
        float e = __expf(-v);
        out[ebase + tid] = __fdividef(1.0f, 1.0f + e);
    }
}

// ---------------------------------------------------------------------------
extern "C" void kernel_launch(void* const* d_in, const int* in_sizes, int n_in,
                              void* d_out, int out_size, void* d_ws, size_t ws_size,
                              hipStream_t stream) {
    const float* z    = (const float*)d_in[0];
    const int*   edge = (const int*)  d_in[1];
    const float* W1   = (const float*)d_in[2];
    const float* b1   = (const float*)d_in[3];
    const float* W2   = (const float*)d_in[4];
    const float* b2   = (const float*)d_in[5];
    float*       out  = (float*)d_out;

    unsigned int* w1bf_u32 = (unsigned int*)d_ws;  // 256*128*2 = 64 KB bf16

    w1_to_bf16_kernel<<<((D_HID * D_IN) / 2 + 255) / 256, 256, 0, stream>>>(W1, w1bf_u32);

    const int nTiles = N_EDGES / 16;               // 500000 / 16 = 31250
    edge_decoder_kernel<<<nTiles, 128, 0, stream>>>(
        z, edge, (const unsigned short*)w1bf_u32, b1, W2, b2, out);
}
